// TransformerMultiheadAttention_24610162606375
// MI455X (gfx1250) — matrix operand order repaired and checked by reading, not compiled, not run on hardware
//
#include <hip/hip_runtime.h>
#include <hip/hip_bf16.h>

// ---------------------------------------------------------------------------
// MHA block on gfx1250 (MI455X): bf16 WMMA GEMMs + flash attention.
// Workspace layout (needs 32 MiB of d_ws):
//   Qbf  : B*S*D   bf16  (8 MiB)   pre-scaled by 1/sqrt(HD) (exact, pow2)
//   KVbf : B*S*2D  bf16  (16 MiB)  [K | V] concatenated along last dim
//   AObf : B*S*D   bf16  (8 MiB)   attention output before final projection
//
// Attention computes S^T = K*Q^T so that (a) softmax reductions are per-lane
// element trees + one shfl_xor(16), and (b) the score C-layout coincides with
// the A-operand layout of P for the P*V WMMA (no LDS re-layout bounce).
// K tiles are staged with GLOBAL_LOAD_ASYNC_TO_LDS_B128 (ASYNCcnt).
// ---------------------------------------------------------------------------

typedef __bf16          v16bf __attribute__((ext_vector_type(16)));
typedef unsigned short  v16u  __attribute__((ext_vector_type(16)));
typedef unsigned short  v8u   __attribute__((ext_vector_type(8)));
typedef float           v8f   __attribute__((ext_vector_type(8)));

#define DEV __device__ __forceinline__

constexpr int Bn = 2, Sn = 2048, Dn = 1024, Hn = 16, HDn = 64;

DEV unsigned short f2bf(float f) {
  unsigned int u = __float_as_uint(f);
  u += 0x7FFFu + ((u >> 16) & 1u);          // round-to-nearest-even
  return (unsigned short)(u >> 16);
}

DEV v16bf as_bf(v16u x) { union { v16u u; v16bf b; } c; c.u = x; return c.b; }

// Assemble an A-operand lane vector from two contiguous 8-element runs
// (ISA 7.12.2: A 16x32 bf16 -> per lane two 8-K runs selected by lane half).
DEV v16bf pack2(v8u lo, v8u hi) {
  union { v16u u; v16bf b; } c;
#pragma unroll
  for (int i = 0; i < 8; ++i) { c.u[i] = lo[i]; c.u[i + 8] = hi[i]; }
  return c.b;
}

DEV v8f wmma_bf16(v16bf a, v16bf b, v8f c) {
  // D = A(16x32 bf16) * B(32x16 bf16) + C(16x16 f32)
  return __builtin_amdgcn_wmma_f32_16x16x32_bf16(
      /*neg_a=*/false, a, /*neg_b=*/false, b,
      /*c_mod=*/(short)0, c, /*reuse_a=*/false, /*reuse_b=*/false);
}

DEV v8u loadA8(const float* p) {
  v8u r;
#pragma unroll
  for (int i = 0; i < 8; ++i) r[i] = f2bf(p[i]);
  return r;
}
DEV v8u loadA8(const unsigned short* p) { return *(const v8u*)p; }

// Async 16-byte global -> LDS copy (per-lane addresses, GV mode, ASYNCcnt).
// Generic pointers to LDS carry the LDS offset in their low 32 bits.
DEV void async_copy_b128(const void* gsrc, void* ldst) {
  unsigned lds = (unsigned)(size_t)ldst;
  unsigned long long g = (unsigned long long)(size_t)gsrc;
  asm volatile("global_load_async_to_lds_b128 %0, %1, off"
               :: "v"(lds), "v"(g) : "memory");
}
DEV void async_wait0() {
  asm volatile("s_wait_asynccnt 0" ::: "memory");
}

// ---------------------------------------------------------------------------
// GEMM: C[M,N] = (A[M,K] * W[K,N] + bias) * oscale.  A: f32 or bf16 (to bf16),
// W: f32 (staged to LDS transposed as bf16), OT: bf16 (ws) or f32 (d_out).
// Block: 256 thr (8 waves), tile 64(M) x 128(N); wave tile 16 x 64.
// ---------------------------------------------------------------------------
template <typename AT, typename OT>
__global__ __launch_bounds__(256) void gemm_wmma(
    const AT* __restrict__ A, const float* __restrict__ W,
    const float* __restrict__ bias, OT* __restrict__ C,
    int M, int N, int K, float oscale) {
  __shared__ __attribute__((aligned(32))) unsigned short LWt[128 * 32]; // [col][k]

  const int tid = threadIdx.x;
  const int wid = tid >> 5, lane = tid & 31;
  const int half = lane >> 4, lrow = lane & 15;
  const int waveM = wid & 3, waveN = wid >> 2;
  const int mb = blockIdx.y * 64, nb = blockIdx.x * 128;

  const AT* Arow = A + (size_t)(mb + waveM * 16 + lrow) * K;

  v8f acc[4] = {};
  for (int k0 = 0; k0 < K; k0 += 32) {
    // Stage W tile (32k x 128n) into LDS, transposed, as bf16.
#pragma unroll
    for (int it = 0; it < 4; ++it) {
      int idx = tid + it * 256;          // 0..1023
      int k   = idx >> 5;                // 0..31
      int c4  = (idx & 31) << 2;         // 0..124
      const float* wp = W + (size_t)(k0 + k) * N + nb + c4;
      const float4 w4 = *(const float4*)wp;
      if (k0 + 32 < K) __builtin_prefetch(wp + (size_t)32 * N, 0, 1);
      LWt[(c4 + 0) * 32 + k] = f2bf(w4.x);
      LWt[(c4 + 1) * 32 + k] = f2bf(w4.y);
      LWt[(c4 + 2) * 32 + k] = f2bf(w4.z);
      LWt[(c4 + 3) * 32 + k] = f2bf(w4.w);
    }
    __syncthreads();

    // A operand: row = lane&15; runs at k0+16*half and k0+16*half+8.
    v16bf a = pack2(loadA8(Arow + k0 + 16 * half),
                    loadA8(Arow + k0 + 16 * half + 8));
#pragma unroll
    for (int ct = 0; ct < 4; ++ct) {
      v16u bv = *(const v16u*)&LWt[(waveN * 64 + ct * 16 + lrow) * 32 + 16 * half];
      acc[ct] = wmma_bf16(a, as_bf(bv), acc[ct]);
    }
    __syncthreads();
  }

#pragma unroll
  for (int ct = 0; ct < 4; ++ct) {
    const int coln = nb + waveN * 64 + ct * 16 + lrow;
    const float bv = bias[coln];
#pragma unroll
    for (int r = 0; r < 8; ++r) {
      const int rown = mb + waveM * 16 + r + 8 * half;   // C layout: row r + 8*half
      float v = (acc[ct][r] + bv) * oscale;
      if constexpr (sizeof(OT) == 2)
        C[(size_t)rown * N + coln] = (OT)f2bf(v);
      else
        C[(size_t)rown * N + coln] = (OT)v;
    }
  }
}

// ---------------------------------------------------------------------------
// Flash attention: grid (S/128, H, B), block 256 (8 waves).
// Each wave owns a 16-row Q tile; K/V streamed in 32-key tiles via LDS.
// Scores computed transposed (S^T = K*Q^T): lane = query, element = key.
// Causal mask with reference semantics (masked score = -10000 before softmax;
// Q pre-scaled by 1/sqrt(HD) at projection time).
// ---------------------------------------------------------------------------
__global__ __launch_bounds__(256) void flash_attn(
    const unsigned short* __restrict__ Qbf,
    const unsigned short* __restrict__ KVbf,
    unsigned short* __restrict__ AObf) {
  __shared__ __attribute__((aligned(32))) unsigned short LK[32 * 64];   // [key][hd]
  __shared__ __attribute__((aligned(32))) unsigned short LVt[64 * 32];  // [hd][key]

  const int tid = threadIdx.x;
  const int wid = tid >> 5, lane = tid & 31;
  const int half = lane >> 4, lrow = lane & 15;
  const int b = blockIdx.z, h = blockIdx.y;
  const int qb = blockIdx.x * 128;
  const int q0 = qb + wid * 16;
  const int qg = q0 + lrow;                  // this lane's query row

  // Q as B-operand (lane = query column): two hd chunks, runs at 8*half and 16+8*half.
  const unsigned short* qrow = Qbf + ((size_t)(b * Sn + qg) * Dn + h * HDn);
  v16bf qb0 = pack2(*(const v8u*)(qrow + 8 * half), *(const v8u*)(qrow + 16 + 8 * half));        // hd [0,32)
  v16bf qb1 = pack2(*(const v8u*)(qrow + 32 + 8 * half), *(const v8u*)(qrow + 48 + 8 * half));   // hd [32,64)

  float m = -1e30f, l = 0.f;                 // per-query running stats (per lane)
  v8f o0 = {}, o1 = {}, o2 = {}, o3 = {};

  const int jend = qb + 128;                 // causal: keys <= last row of block
  for (int j0 = 0; j0 < jend; j0 += 32) {
    // Stage tiles: K via async global->LDS copy; V via transposed scatter.
    {
      const int row = tid >> 3;              // 0..31 (key within tile)
      const int off = (tid & 7) << 3;        // 0,8,...,56 (hd)
      const unsigned short* src =
          KVbf + ((size_t)(b * Sn + j0 + row) * (2 * Dn) + h * HDn);
      async_copy_b128(src + off, &LK[row * 64 + off]);
      v8u vv = *(const v8u*)(src + Dn + off);
#pragma unroll
      for (int i = 0; i < 8; ++i) LVt[(off + i) * 32 + ((row >> 3) & 1) * 16 + (row & 7) + ((row >> 4) << 3)] = vv[i];
      if (j0 + 32 < jend) {                  // prefetch next K/V tile
        __builtin_prefetch(src + (size_t)32 * 2 * Dn + off, 0, 1);
        __builtin_prefetch(src + (size_t)32 * 2 * Dn + Dn + off, 0, 1);
      }
    }
    async_wait0();
    __syncthreads();

    // S^T tiles: st0 = keys [j0,j0+16), st1 = keys [j0+16,j0+32).
    // A = K rows from LDS, B = Q (registers); K-reduction over HD=64 chained.
    v16bf ka;
    v8f st0 = {}, st1 = {};
    ka  = pack2(*(const v8u*)&LK[lrow * 64 + 8 * half],
                *(const v8u*)&LK[lrow * 64 + 16 + 8 * half]);
    st0 = wmma_bf16(ka, qb0, st0);
    ka  = pack2(*(const v8u*)&LK[lrow * 64 + 32 + 8 * half],
                *(const v8u*)&LK[lrow * 64 + 48 + 8 * half]);
    st0 = wmma_bf16(ka, qb1, st0);
    ka  = pack2(*(const v8u*)&LK[(16 + lrow) * 64 + 8 * half],
                *(const v8u*)&LK[(16 + lrow) * 64 + 16 + 8 * half]);
    st1 = wmma_bf16(ka, qb0, st1);
    ka  = pack2(*(const v8u*)&LK[(16 + lrow) * 64 + 32 + 8 * half],
                *(const v8u*)&LK[(16 + lrow) * 64 + 48 + 8 * half]);
    st1 = wmma_bf16(ka, qb1, st1);

    // Causal mask (element r = key j0 + r + 8*half [+16 for st1]).
    float a0[8], a1[8];
#pragma unroll
    for (int r = 0; r < 8; ++r) {
      const int kgl = j0 + r + 8 * half;
      a0[r] = (kgl      <= qg) ? st0[r] : -10000.f;
      a1[r] = (kgl + 16 <= qg) ? st1[r] : -10000.f;
    }

    // Per-lane max over 16 elements, then merge lane halves (full 32 keys).
    float mx = fmaxf(a0[0], a1[0]);
#pragma unroll
    for (int r = 1; r < 8; ++r) mx = fmaxf(mx, fmaxf(a0[r], a1[r]));
    mx = fmaxf(mx, __shfl_xor(mx, 16, 32));

    const float mn  = fmaxf(m, mx);
    const float fac = __expf(m - mn);
    m = mn;

    float p0[8], p1[8], sum = 0.f;
#pragma unroll
    for (int r = 0; r < 8; ++r) {
      p0[r] = __expf(a0[r] - mn);
      p1[r] = __expf(a1[r] - mn);
      sum += p0[r] + p1[r];
    }
    sum += __shfl_xor(sum, 16, 32);
    l = l * fac + sum;

    // P in A-operand layout directly (runs {0..7,16..23}/{8..15,24..31}).
    union { v16u u; v16bf bf; } pa;
#pragma unroll
    for (int r = 0; r < 8; ++r) { pa.u[r] = f2bf(p0[r]); pa.u[r + 8] = f2bf(p1[r]); }

    // Rescale O: factor for query row (r + 8*half) lives in lane (r + 8*half).
#pragma unroll
    for (int r = 0; r < 8; ++r) {
      const float fr = __shfl(fac, r + 8 * half, 32);
      o0[r] *= fr; o1[r] *= fr; o2[r] *= fr; o3[r] *= fr;
    }

    // O += P * V  (B = V^T tile from LDS, lane = hd column).
    o0 = wmma_bf16(pa.bf, as_bf(*(const v16u*)&LVt[( 0 + lrow) * 32 + 16 * half]), o0);
    o1 = wmma_bf16(pa.bf, as_bf(*(const v16u*)&LVt[(16 + lrow) * 32 + 16 * half]), o1);
    o2 = wmma_bf16(pa.bf, as_bf(*(const v16u*)&LVt[(32 + lrow) * 32 + 16 * half]), o2);
    o3 = wmma_bf16(pa.bf, as_bf(*(const v16u*)&LVt[(48 + lrow) * 32 + 16 * half]), o3);
    __syncthreads();
  }

  // Normalize and store (O layout: lane = hd col, element r = query r+8*half).
#pragma unroll
  for (int r = 0; r < 8; ++r) {
    const float lrq = __shfl(l, r + 8 * half, 32);
    const float inv = 1.f / lrq;
    unsigned short* dst =
        AObf + ((size_t)(b * Sn + q0 + r + 8 * half) * Dn + h * HDn + lrow);
    dst[0]  = f2bf(o0[r] * inv);
    dst[16] = f2bf(o1[r] * inv);
    dst[32] = f2bf(o2[r] * inv);
    dst[48] = f2bf(o3[r] * inv);
  }
}

// ---------------------------------------------------------------------------

extern "C" void kernel_launch(void* const* d_in, const int* in_sizes, int n_in,
                              void* d_out, int out_size, void* d_ws, size_t ws_size,
                              hipStream_t stream) {
  (void)in_sizes; (void)n_in; (void)out_size; (void)ws_size;
  const float* Xq  = (const float*)d_in[0];
  const float* Xkv = (const float*)d_in[1];
  // d_in[2] is the causal mask; applied analytically in flash_attn.
  const float* Wq  = (const float*)d_in[3];
  const float* bq  = (const float*)d_in[4];
  const float* Wkv = (const float*)d_in[5];
  const float* bkv = (const float*)d_in[6];
  const float* Wp  = (const float*)d_in[7];
  const float* bp  = (const float*)d_in[8];
  float* out = (float*)d_out;

  unsigned short* Qbf  = (unsigned short*)d_ws;
  unsigned short* KVbf = Qbf + (size_t)Bn * Sn * Dn;
  unsigned short* AObf = KVbf + (size_t)Bn * Sn * 2 * Dn;   // total 32 MiB used

  const int M = Bn * Sn;   // 4096
  dim3 blk(256);
  gemm_wmma<float, unsigned short>
      <<<dim3(Dn / 128, M / 64), blk, 0, stream>>>(Xq, Wq, bq, Qbf, M, Dn, Dn,
                                                   0.125f /* 1/sqrt(HD), exact */);
  gemm_wmma<float, unsigned short>
      <<<dim3(2 * Dn / 128, M / 64), blk, 0, stream>>>(Xkv, Wkv, bkv, KVbf, M, 2 * Dn, Dn, 1.0f);
  flash_attn<<<dim3(Sn / 128, Hn, Bn), blk, 0, stream>>>(Qbf, KVbf, AObf);
  gemm_wmma<unsigned short, float>
      <<<dim3(Dn / 128, M / 64), blk, 0, stream>>>(AObf, Wp, bp, out, M, Dn, Dn, 1.0f);
}
